// CentroidHead_72456098283856
// MI455X (gfx1250) — matrix operand order repaired and checked by reading, not compiled, not run on hardware
//
#include <hip/hip_runtime.h>
#include <hip/hip_bf16.h>
#include <math.h>

#define GXD 128
#define GYD 128
#define GZD 128
#define NKEYS (2 * GXD * GYD * GZD)   // NBATCH=2 -> 4,194,304 keys

typedef __attribute__((ext_vector_type(16))) __bf16 v16bf;
typedef __attribute__((ext_vector_type(8)))  float  v8f;
typedef __attribute__((ext_vector_type(8)))  int    v8i_t;

__device__ __forceinline__ unsigned short f2bf(float f) {
  unsigned u = __builtin_bit_cast(unsigned, f);
  u += 0x7fffu + ((u >> 16) & 1u);           // round to nearest even
  return (unsigned short)(u >> 16);
}

__device__ __forceinline__ int nbr_lookup(const int* __restrict__ tbl,
                                          int b, int x, int y, int z) {
  if ((unsigned)x >= GXD || (unsigned)y >= GYD || (unsigned)z >= GZD) return -1;
  int key = ((b * GXD + x) * GYD + y) * GZD + z;
  return tbl[key];
}

// ---------------------------------------------------------------- tables ----
__global__ void k_init_tables(int* __restrict__ t, int n) {
  int i = blockIdx.x * blockDim.x + threadIdx.x;
  if (i < n) t[i] = -1;
}

__global__ void k_zero_u16(unsigned short* __restrict__ p, int n) {
  int i = blockIdx.x * blockDim.x + threadIdx.x;
  if (i < n) p[i] = 0;
}

__global__ void k_scatter(const int* __restrict__ coords, int* __restrict__ tbl, int n) {
  int i = blockIdx.x * blockDim.x + threadIdx.x;
  if (i >= n) return;
  const int* c = coords + (size_t)i * 4;
  int key = ((c[0] * GXD + c[1]) * GYD + c[2]) * GZD + c[3];
  tbl[key] = i;
}

// ------------------------------------------------- x0 = concat(feats, offB) -
// x0: nvox rows x 96 bf16 (65 live channels, zero padded)
__global__ void k_build_x0(const int* __restrict__ vcoords, const float* __restrict__ vfeats,
                           const float* __restrict__ ofeats, const int* __restrict__ tblB,
                           unsigned short* __restrict__ x0, int nvox) {
  int e = blockIdx.x * blockDim.x + threadIdx.x;
  if (e >= nvox * 96) return;
  int v = e / 96, c = e - v * 96;
  float val = 0.f;
  if (c < 64) {
    val = vfeats[(size_t)v * 64 + c];
  } else if (c == 64) {
    const int* cp = vcoords + (size_t)v * 4;
    int key = ((cp[0] * GXD + cp[1]) * GYD + cp[2]) * GZD + cp[3];
    int j = tblB[key];
    val = (j >= 0) ? ofeats[j] : 0.f;
  }
  x0[e] = f2bf(val);
}

// ----------------------------------------- weights -> WMMA B operand layout -
// B (32x16, KxN) as packed here (contiguous k, same order as A): lane l holds N=l&15, K = (l>>4)*16 + h, h=0..15
__global__ void k_pack_w1(const float* __restrict__ W1, unsigned short* __restrict__ W1p) {
  int e = blockIdx.x * blockDim.x + threadIdx.x;
  if (e >= 27 * 3 * 2 * 512) return;
  int h = e & 15, l = (e >> 4) & 31, t = (e >> 9) & 1, c = (e >> 10) % 3, off = e / 3072;
  int k = c * 32 + (l >> 4) * 16 + h;
  int n = (l & 15) + 16 * t;
  float val = (k < 65) ? W1[((size_t)off * 65 + k) * 32 + n] : 0.f;
  W1p[e] = f2bf(val);
}

__global__ void k_pack_w2(const float* __restrict__ W2, unsigned short* __restrict__ W2p) {
  int e = blockIdx.x * blockDim.x + threadIdx.x;
  if (e >= 27 * 512) return;
  int h = e & 15, l = (e >> 4) & 31, off = e >> 9;
  int k = (l >> 4) * 16 + h;
  int n = l & 15;
  W2p[e] = f2bf(W2[((size_t)off * 32 + k) * 16 + n]);
}

// ---------------------------------------------------------------- conv1 -----
// One wave per 16-voxel tile. Double-buffered async-to-LDS gather overlapped
// with 27 offsets x 3 K-chunks x 2 N-tiles of v_wmma_f32_16x16x32_bf16.
__global__ void __launch_bounds__(256) k_conv1(
    const int* __restrict__ coords, const unsigned short* __restrict__ x0,
    const unsigned short* __restrict__ W1p, const int* __restrict__ tblA,
    const unsigned short* __restrict__ zrow,
    unsigned short* __restrict__ x1, int nvox) {
  __shared__ unsigned short lds[8][2][16 * 96];
  const int wv   = threadIdx.x >> 5;
  const int lane = threadIdx.x & 31;
  const int tile = blockIdx.x * 8 + wv;
  const int row0 = tile * 16;
  if (row0 >= nvox) return;
  const int r  = lane & 15;        // row handled by this lane (gather + A-matrix M)
  const int hf = lane >> 4;        // half selector
  const int g  = row0 + r;
  const bool rok = g < nvox;
  int cb = 0, cx = 0, cy = 0, cz = 0;
  if (rok) { const int* cp = coords + (size_t)g * 4; cb = cp[0]; cx = cp[1]; cy = cp[2]; cz = cp[3]; }

  // per-lane LDS byte addresses (flat LDS addr truncated to 32b = LDS offset)
  unsigned lb0 = (unsigned)(unsigned long long)&lds[wv][0][r * 96 + hf * 48];
  unsigned lb1 = (unsigned)(unsigned long long)&lds[wv][1][r * 96 + hf * 48];

  // issue a 96B-per-lane async gather of the neighbor row for `off` into buf
  auto issue_gather = [&](int off, int buf) {
    const int dx = off / 9 - 1, dy = (off / 3) % 3 - 1, dz = off % 3 - 1;
    const int nb = rok ? nbr_lookup(tblA, cb, cx + dx, cy + dy, cz + dz) : -1;
    const unsigned short* src = (nb >= 0) ? &x0[(size_t)nb * 96 + hf * 48]
                                          : (zrow + hf * 48);
    unsigned long long ga = (unsigned long long)src;
    unsigned lb = buf ? lb1 : lb0;
    asm volatile("global_load_async_to_lds_b128 %0, %1, off"            :: "v"(lb), "v"(ga) : "memory");
    asm volatile("global_load_async_to_lds_b128 %0, %1, off offset:16" :: "v"(lb), "v"(ga) : "memory");
    asm volatile("global_load_async_to_lds_b128 %0, %1, off offset:32" :: "v"(lb), "v"(ga) : "memory");
    asm volatile("global_load_async_to_lds_b128 %0, %1, off offset:48" :: "v"(lb), "v"(ga) : "memory");
    asm volatile("global_load_async_to_lds_b128 %0, %1, off offset:64" :: "v"(lb), "v"(ga) : "memory");
    asm volatile("global_load_async_to_lds_b128 %0, %1, off offset:80" :: "v"(lb), "v"(ga) : "memory");
  };

  v8f acc0 = {0.f, 0.f, 0.f, 0.f, 0.f, 0.f, 0.f, 0.f};
  v8f acc1 = {0.f, 0.f, 0.f, 0.f, 0.f, 0.f, 0.f, 0.f};

  issue_gather(0, 0);
  for (int off = 0; off < 27; ++off) {
    const int buf = off & 1;
    if (off < 26) {
      issue_gather(off + 1, buf ^ 1);
      asm volatile("s_wait_asynccnt 0x6" ::: "memory");  // prev 6 (in-order) done
    } else {
      asm volatile("s_wait_asynccnt 0x0" ::: "memory");
    }
    const unsigned short* L = &lds[wv][buf][0];
    #pragma unroll
    for (int c = 0; c < 3; ++c) {
      // assemble A operand per 16-bit 16x32 wave32 layout
      v8i_t ra;
      #pragma unroll
      for (int v = 0; v < 8; ++v) {
        const int kl = 2 * v + hf * 16;
        ra[v] = *(const int*)&L[r * 96 + c * 32 + kl];
      }
      const v16bf A = __builtin_bit_cast(v16bf, ra);
      const size_t bbase = ((((size_t)off * 3 + c) * 2) * 32 + lane) * 16;
      const v16bf B0 = __builtin_bit_cast(v16bf, *(const v8i_t*)&W1p[bbase]);
      const v16bf B1 = __builtin_bit_cast(v16bf, *(const v8i_t*)&W1p[bbase + 512]);
      acc0 = __builtin_amdgcn_wmma_f32_16x16x32_bf16(false, A, false, B0, (short)0, acc0, false, false);
      acc1 = __builtin_amdgcn_wmma_f32_16x16x32_bf16(false, A, false, B1, (short)0, acc1, false, false);
    }
  }
  // C/D layout: lane l VGPR j -> M = j + 8*(l>>4), N = l&15 (+16 for acc1)
  const int n0 = lane & 15;
  #pragma unroll
  for (int j = 0; j < 8; ++j) {
    const int m = j + 8 * hf;
    const int gr = row0 + m;
    if (gr < nvox) {
      float a = acc0[j]; a = a > 0.f ? a : 0.f;
      float b = acc1[j]; b = b > 0.f ? b : 0.f;
      x1[(size_t)gr * 32 + n0]      = f2bf(a);
      x1[(size_t)gr * 32 + n0 + 16] = f2bf(b);
    }
  }
}

// -------------------------------------------- conv2 + W3 dot + sigmoid ------
__global__ void __launch_bounds__(256) k_conv2(
    const int* __restrict__ coords, const unsigned short* __restrict__ x1,
    const unsigned short* __restrict__ W2p, const int* __restrict__ tblA,
    const unsigned short* __restrict__ zrow,
    const float* __restrict__ W3, const float* __restrict__ b3,
    float* __restrict__ sc_out, float* __restrict__ s_ws, int nvox) {
  __shared__ unsigned short lds[8][2][16 * 32];
  const int wv   = threadIdx.x >> 5;
  const int lane = threadIdx.x & 31;
  const int tile = blockIdx.x * 8 + wv;
  const int row0 = tile * 16;
  if (row0 >= nvox) return;
  const int r  = lane & 15;
  const int hf = lane >> 4;
  const int g  = row0 + r;
  const bool rok = g < nvox;
  int cb = 0, cx = 0, cy = 0, cz = 0;
  if (rok) { const int* cp = coords + (size_t)g * 4; cb = cp[0]; cx = cp[1]; cy = cp[2]; cz = cp[3]; }

  unsigned lb0 = (unsigned)(unsigned long long)&lds[wv][0][r * 32 + hf * 16];
  unsigned lb1 = (unsigned)(unsigned long long)&lds[wv][1][r * 32 + hf * 16];

  auto issue_gather = [&](int off, int buf) {
    const int dx = off / 9 - 1, dy = (off / 3) % 3 - 1, dz = off % 3 - 1;
    const int nb = rok ? nbr_lookup(tblA, cb, cx + dx, cy + dy, cz + dz) : -1;
    const unsigned short* src = (nb >= 0) ? &x1[(size_t)nb * 32 + hf * 16]
                                          : (zrow + hf * 16);
    unsigned long long ga = (unsigned long long)src;
    unsigned lb = buf ? lb1 : lb0;
    asm volatile("global_load_async_to_lds_b128 %0, %1, off"            :: "v"(lb), "v"(ga) : "memory");
    asm volatile("global_load_async_to_lds_b128 %0, %1, off offset:16" :: "v"(lb), "v"(ga) : "memory");
  };

  v8f acc = {0.f, 0.f, 0.f, 0.f, 0.f, 0.f, 0.f, 0.f};

  issue_gather(0, 0);
  for (int off = 0; off < 27; ++off) {
    const int buf = off & 1;
    if (off < 26) {
      issue_gather(off + 1, buf ^ 1);
      asm volatile("s_wait_asynccnt 0x2" ::: "memory");
    } else {
      asm volatile("s_wait_asynccnt 0x0" ::: "memory");
    }
    const unsigned short* L = &lds[wv][buf][0];
    v8i_t ra;
    #pragma unroll
    for (int v = 0; v < 8; ++v) {
      const int kl = 2 * v + hf * 16;
      ra[v] = *(const int*)&L[r * 32 + kl];
    }
    const v16bf A = __builtin_bit_cast(v16bf, ra);
    const v16bf B = __builtin_bit_cast(v16bf, *(const v8i_t*)&W2p[((size_t)off * 32 + lane) * 16]);
    acc = __builtin_amdgcn_wmma_f32_16x16x32_bf16(false, A, false, B, (short)0, acc, false, false);
  }
  // fused relu -> dot W3 (reduce over N across 16-lane group) -> sigmoid
  const float w3n = W3[lane & 15];
  const float bb  = b3[0];
  #pragma unroll
  for (int j = 0; j < 8; ++j) {
    float a = acc[j]; a = a > 0.f ? a : 0.f;
    float p = a * w3n;
    p += __shfl_xor(p, 1);
    p += __shfl_xor(p, 2);
    p += __shfl_xor(p, 4);
    p += __shfl_xor(p, 8);
    const int m = j + 8 * hf;
    const int gr = row0 + m;
    if ((lane & 15) == 0 && gr < nvox) {
      float sv = 1.f / (1.f + __expf(-(p + bb)));
      sc_out[gr] = sv;
      s_ws[gr]   = sv;
    }
  }
}

// --------------------------------------------------------------- peaks ------
__global__ void k_peak(const int* __restrict__ coords, const int* __restrict__ tblA,
                       const float* __restrict__ s, float* __restrict__ stmp, int nvox) {
  int i = blockIdx.x * blockDim.x + threadIdx.x;
  if (i >= nvox) return;
  const float si = s[i];
  const bool mask = si > 0.1f;
  float hmax = -INFINITY;
  const int* cp = coords + (size_t)i * 4;
  const int b = cp[0], x = cp[1], y = cp[2], z = cp[3];
  for (int o = 0; o < 27; ++o) {
    int nb = nbr_lookup(tblA, b, x + o / 9 - 1, y + (o / 3) % 3 - 1, z + o % 3 - 1);
    if (nb >= 0) {
      float sv = s[nb];
      if (sv > 0.1f && sv > hmax) hmax = sv;
    }
  }
  const bool peak = mask && (hmax == si) && (si > 0.2f);
  stmp[i] = peak ? si : -INFINITY;
}

// --------------------------------------------------------- cluster avg ------
__global__ void k_cluster(const int* __restrict__ ccoords, const float* __restrict__ cfeats,
                          const int* __restrict__ tblC, float* __restrict__ cavg, int nclu) {
  int t = blockIdx.x * blockDim.x + threadIdx.x;
  if (t >= nclu * 4) return;
  const int ci = t >> 2, ch = (t & 3) * 16;
  const int* cp = ccoords + (size_t)ci * 4;
  const int b = cp[0], x = cp[1], y = cp[2], z = cp[3];
  float acc[16];
  #pragma unroll
  for (int q = 0; q < 16; ++q) acc[q] = 0.f;
  float cnt = 0.f;
  for (int o = 0; o < 27; ++o) {
    int nb = nbr_lookup(tblC, b, x + o / 9 - 1, y + (o / 3) % 3 - 1, z + o % 3 - 1);
    if (nb >= 0) {
      cnt += 1.f;
      const float* f = &cfeats[(size_t)nb * 64 + ch];
      #pragma unroll
      for (int q = 0; q < 16; ++q) acc[q] += f[q];
    }
  }
  const float inv = 1.f / cnt;
  float* o = &cavg[(size_t)ci * 64 + ch];
  #pragma unroll
  for (int q = 0; q < 16; ++q) o[q] = acc[q] * inv;
}

// --------------------------------------------------------------- top-k ------
// Deterministic: 128 passes of argmax with lowest-index tie-break (matches top_k).
__global__ void __launch_bounds__(1024) k_topk(float* __restrict__ stmp, int n,
                                               int* __restrict__ topi, float* __restrict__ topv) {
  __shared__ float sv[1024];
  __shared__ int   si[1024];
  for (int rr = 0; rr < 128; ++rr) {
    float best = -INFINITY;
    int bi = 0x7fffffff;
    for (int i = threadIdx.x; i < n; i += 1024) {
      float v = stmp[i];
      if (v > best || (v == best && i < bi)) { best = v; bi = i; }
    }
    sv[threadIdx.x] = best; si[threadIdx.x] = bi;
    __syncthreads();
    for (int s = 512; s > 0; s >>= 1) {
      if (threadIdx.x < (unsigned)s) {
        float ov = sv[threadIdx.x + s]; int oi = si[threadIdx.x + s];
        if (ov > sv[threadIdx.x] || (ov == sv[threadIdx.x] && oi < si[threadIdx.x])) {
          sv[threadIdx.x] = ov; si[threadIdx.x] = oi;
        }
      }
      __syncthreads();
    }
    if (threadIdx.x == 0) {
      topi[rr] = si[0];
      topv[rr] = sv[0];
      stmp[si[0]] = -INFINITY;
    }
    __syncthreads();
  }
}

// ------------------------------------------------------------- finalize -----
// out layout (after scores): coords 128x4 | feats 128x64 | peak_scores 128
__global__ void k_final(const int* __restrict__ vcoords, const float* __restrict__ cavg,
                        const int* __restrict__ inv_map, const int* __restrict__ topi,
                        const float* __restrict__ topv, float* __restrict__ out) {
  int e = blockIdx.x * blockDim.x + threadIdx.x;
  if (e >= 128 * 64) return;
  const int r = e >> 6, c = e & 63;
  const int vi = topi[r];
  float* out_coords = out;
  float* out_feats  = out + 512;
  float* out_scores = out + 512 + 8192;
  out_feats[e] = cavg[(size_t)inv_map[vi] * 64 + c];
  if (c < 4)  out_coords[r * 4 + c] = (float)vcoords[(size_t)vi * 4 + c];
  if (c == 0) out_scores[r] = topv[r];
}

// ---------------------------------------------------------------------------
extern "C" void kernel_launch(void* const* d_in, const int* in_sizes, int n_in,
                              void* d_out, int out_size, void* d_ws, size_t ws_size,
                              hipStream_t stream) {
  const int*   vcoords = (const int*)d_in[0];
  const float* vfeats  = (const float*)d_in[1];
  const int*   ocoords = (const int*)d_in[2];
  const float* ofeats  = (const float*)d_in[3];
  const int*   ccoords = (const int*)d_in[4];
  const float* cfeats  = (const float*)d_in[5];
  const int*   inv_map = (const int*)d_in[6];
  const float* W1      = (const float*)d_in[7];
  const float* W2      = (const float*)d_in[8];
  const float* W3      = (const float*)d_in[9];
  const float* b3      = (const float*)d_in[10];
  const int nvox = in_sizes[0] / 4;
  const int noff = in_sizes[2] / 4;
  const int nclu = in_sizes[4] / 4;

  char* ws = (char*)d_ws;
  size_t cur = 0;
  auto alloc = [&](size_t bytes) -> void* {
    void* p = ws + cur;
    cur += (bytes + 255) & ~(size_t)255;
    return p;
  };
  int* tblA = (int*)alloc((size_t)NKEYS * 4);
  int* tblB = (int*)alloc((size_t)NKEYS * 4);
  int* tblC = (int*)alloc((size_t)NKEYS * 4);
  unsigned short* x0   = (unsigned short*)alloc((size_t)nvox * 96 * 2);
  unsigned short* x1   = (unsigned short*)alloc((size_t)nvox * 32 * 2);
  unsigned short* W1p  = (unsigned short*)alloc((size_t)27 * 3 * 2 * 512 * 2);
  unsigned short* W2p  = (unsigned short*)alloc((size_t)27 * 512 * 2);
  unsigned short* zrow = (unsigned short*)alloc(96 * 2);   // zero neighbor row
  float* s    = (float*)alloc((size_t)nvox * 4);
  float* stmp = (float*)alloc((size_t)nvox * 4);
  float* cavg = (float*)alloc((size_t)nclu * 64 * 4);
  int*   topi = (int*)alloc(128 * 4);
  float* topv = (float*)alloc(128 * 4);

  const int B = 256;
  const int ninit = 3 * NKEYS;   // tblA/B/C are contiguous
  k_init_tables<<<(ninit + B - 1) / B, B, 0, stream>>>(tblA, ninit);
  k_zero_u16<<<1, 128, 0, stream>>>(zrow, 96);
  k_scatter<<<(nvox + B - 1) / B, B, 0, stream>>>(vcoords, tblA, nvox);
  k_scatter<<<(noff + B - 1) / B, B, 0, stream>>>(ocoords, tblB, noff);
  k_scatter<<<(nclu + B - 1) / B, B, 0, stream>>>(ccoords, tblC, nclu);
  k_build_x0<<<((size_t)nvox * 96 + B - 1) / B, B, 0, stream>>>(vcoords, vfeats, ofeats, tblB, x0, nvox);
  k_pack_w1<<<(27 * 3 * 2 * 512 + B - 1) / B, B, 0, stream>>>(W1, W1p);
  k_pack_w2<<<(27 * 512 + B - 1) / B, B, 0, stream>>>(W2, W2p);

  const int tiles = (nvox + 15) / 16;
  k_conv1<<<(tiles + 7) / 8, 256, 0, stream>>>(vcoords, x0, W1p, tblA, zrow, x1, nvox);
  k_conv2<<<(tiles + 7) / 8, 256, 0, stream>>>(vcoords, x1, W2p, tblA, zrow, W3, b3,
                                               (float*)d_out, s, nvox);
  k_peak<<<(nvox + B - 1) / B, B, 0, stream>>>(vcoords, tblA, s, stmp, nvox);
  k_cluster<<<(nclu * 4 + B - 1) / B, B, 0, stream>>>(ccoords, cfeats, tblC, cavg, nclu);
  k_topk<<<1, 1024, 0, stream>>>(stmp, nvox, topi, topv);
  k_final<<<(128 * 64 + B - 1) / B, B, 0, stream>>>(vcoords, cavg, inv_map, topi, topv,
                                                    (float*)d_out + nvox);
}
